// Attention_24927990186341
// MI455X (gfx1250) — matrix operand order repaired and checked by reading, not compiled, not run on hardware
//
#include <hip/hip_runtime.h>
#include <hip/hip_bf16.h>
#include <cstddef>
#include <cstdint>

// ---------------------------------------------------------------------------
// Perceiver cross-attention for MI455X (gfx1250, wave32, WMMA + TDM).
//
// Pipeline (bf16 storage, f32 accumulation):
//   K0: Wq/Wkv/Wout  f32[K][N] -> bf16 [N][K] (transposed for B-frag loads)
//   K1: LayerNorm(x rows | latent rows) -> xn bf16 [8][4160][1024]
//   K2: kv  = xn @ Wkv   (WMMA GEMM, TDM double-buffered LDS, bf16 out)
//   K3: q   = ln @ Wq    (same GEMM)
//   K4: flash attention per (b,h): K tile via TDM, sim & P@V via WMMA
//   K5: out = attn @ Wout + bout (same GEMM, f32 out + bias)
// ---------------------------------------------------------------------------

typedef __bf16 v16bf __attribute__((ext_vector_type(16)));
typedef __bf16 v8bf  __attribute__((ext_vector_type(8)));
typedef float  v8f   __attribute__((ext_vector_type(8)));
typedef unsigned int v4u  __attribute__((ext_vector_type(4)));
typedef int          v8i_ __attribute__((ext_vector_type(8)));
typedef int          v4i_ __attribute__((ext_vector_type(4)));

static constexpr int B_   = 8;
static constexpr int F_   = 4096;
static constexpr int Q_   = 64;
static constexpr int FQ   = F_ + Q_;      // 4160
static constexpr int DIMC = 1024;
static constexpr int HID  = 1024;
static constexpr int NH   = 16;
static constexpr int DH   = 64;

// --------------------------- TDM support -----------------------------------
#if defined(__HIP_DEVICE_COMPILE__) && defined(__gfx1250__) && \
    __has_builtin(__builtin_amdgcn_tensor_load_to_lds)
#define USE_TDM 1
#if __has_include(<hip/amd_detail/amd_gfx1250_TDM.h>)
#define TDM_6ARG 1
#endif
#endif

// Double-buffer index: fallback (synchronous-load) path reuses buffer 0.
__device__ __forceinline__ constexpr int dbuf(int b) {
#if defined(USE_TDM)
  return b;
#else
  return 0;
#endif
}

// Low 32 bits of a generic pointer to __shared__ = LDS byte address
// (flat LDS aperture: the low 32 address bits are the LDS address).
__device__ __forceinline__ unsigned ldsAddr(const void* p) {
  return (unsigned)(unsigned long long)(uintptr_t)p;
}

#if defined(USE_TDM)
// 2D tile load: tile[rows][cols] of bf16 from global (row stride in elements)
// packed tight into LDS at lds_addr.  D# built per CDNA5 ISA 8.3/8.4.
__device__ __forceinline__ void tdm_load_2d(unsigned lds_addr, const void* gptr,
                                            unsigned tile_cols, unsigned tile_rows,
                                            unsigned long long row_stride_elems) {
  const unsigned long long ga = (unsigned long long)(uintptr_t)gptr;
  v4u g0;
  g0[0] = 1u;                                        // count=1, user mode
  g0[1] = lds_addr;                                  // LDS byte address
  g0[2] = (unsigned)(ga & 0xffffffffu);              // global_addr[31:0]
  g0[3] = (unsigned)((ga >> 32) & 0x01ffffffu)       // global_addr[56:32]
          | (2u << 30);                              // type = 2 ("image")
  v8i_ g1;
  g1[0] = (int)(1u << 16);                           // data_size=1 (2 bytes)
  g1[1] = (int)((tile_cols & 0xffffu) << 16);        // tensor_dim0[15:0]
  g1[2] = (int)(((tile_cols >> 16) & 0xffffu) |      // tensor_dim0[31:16]
                ((tile_rows & 0xffffu) << 16));      // tensor_dim1[15:0]
  g1[3] = (int)(((tile_rows >> 16) & 0xffffu) |      // tensor_dim1[31:16]
                ((tile_cols & 0xffffu) << 16));      // tile_dim0
  g1[4] = (int)(tile_rows & 0xffffu);                // tile_dim1 (tile_dim2=0)
  g1[5] = (int)(unsigned)(row_stride_elems & 0xffffffffu);     // dim0_stride lo
  g1[6] = (int)(unsigned)((row_stride_elems >> 32) & 0xffffu); // dim0_stride hi
  g1[7] = 0;                                         // tensor_dim1_stride = 0
  v4i_ z4 = {0, 0, 0, 0};
#if defined(TDM_6ARG)
  v8i_ z8 = {0, 0, 0, 0, 0, 0, 0, 0};
  __builtin_amdgcn_tensor_load_to_lds(g0, g1, z4, z4, z8, 0);
#else
  __builtin_amdgcn_tensor_load_to_lds(g0, g1, z4, z4, 0);
#endif
}
#endif  // USE_TDM

// Combine two 16B chunks into one 16-element bf16 fragment register.
__device__ __forceinline__ v16bf ld_frag2(const __bf16* p0, const __bf16* p1) {
  v8bf a = *(const v8bf*)p0;
  v8bf b = *(const v8bf*)p1;
  v16bf r;
#pragma unroll
  for (int i = 0; i < 8; ++i) { r[i] = a[i]; r[i + 8] = b[i]; }
  return r;
}

// ---------------------------------------------------------------------------
// K0: weight transpose + f32 -> bf16.  W: [K][N] f32  ->  Wt: [N][K] bf16
// ---------------------------------------------------------------------------
__global__ __launch_bounds__(256) void wtrans_kernel(const float* __restrict__ W,
                                                     __bf16* __restrict__ Wt,
                                                     int N, int K) {
  __shared__ float t[32][33];
  const int n0 = blockIdx.x * 32, k0 = blockIdx.y * 32;
  const int tx = threadIdx.x, ty = threadIdx.y;
#pragma unroll
  for (int i = 0; i < 32; i += 8)
    t[ty + i][tx] = W[(size_t)(k0 + ty + i) * N + n0 + tx];
  __syncthreads();
#pragma unroll
  for (int i = 0; i < 32; i += 8)
    Wt[(size_t)(n0 + ty + i) * K + k0 + tx] = (__bf16)t[tx][ty + i];
}

// ---------------------------------------------------------------------------
// K1: LayerNorm. One block per row (8*4160 rows of 1024).
// ---------------------------------------------------------------------------
__global__ __launch_bounds__(256) void ln_kernel(const float* __restrict__ x,
                                                 const float* __restrict__ lat,
                                                 const float* __restrict__ gm,
                                                 const float* __restrict__ bm,
                                                 const float* __restrict__ gl,
                                                 const float* __restrict__ bl,
                                                 __bf16* __restrict__ xn) {
  const int row = blockIdx.x;               // 0..33279
  const int b = row / FQ, f = row % FQ;
  const float* src;
  const float* g;
  const float* be;
  if (f < F_) { src = x   + ((size_t)b * F_ + f)        * DIMC; g = gm; be = bm; }
  else        { src = lat + ((size_t)b * Q_ + (f - F_)) * DIMC; g = gl; be = bl; }

  const int tid = threadIdx.x;
  float v[4], s = 0.f, ss = 0.f;
#pragma unroll
  for (int i = 0; i < 4; ++i) {
    v[i] = src[tid + i * 256];
    s += v[i];
    ss += v[i] * v[i];
  }
#pragma unroll
  for (int d = 16; d > 0; d >>= 1) {
    s  += __shfl_xor(s,  d, 32);
    ss += __shfl_xor(ss, d, 32);
  }
  __shared__ float sh[16];
  const int w = tid >> 5, lane = tid & 31;
  if (lane == 0) { sh[w] = s; sh[w + 8] = ss; }
  __syncthreads();
  if (tid == 0) {
    float ts = 0.f, tss = 0.f;
#pragma unroll
    for (int i = 0; i < 8; ++i) { ts += sh[i]; tss += sh[i + 8]; }
    sh[0] = ts; sh[1] = tss;
  }
  __syncthreads();
  const float mean = sh[0] * (1.f / 1024.f);
  const float var  = sh[1] * (1.f / 1024.f) - mean * mean;
  const float rstd = rsqrtf(var + 1e-5f);
  __bf16* dst = xn + (size_t)row * DIMC;
#pragma unroll
  for (int i = 0; i < 4; ++i) {
    const int idx = tid + i * 256;
    dst[idx] = (__bf16)((v[i] - mean) * rstd * g[idx] + be[idx]);
  }
}

// ---------------------------------------------------------------------------
// Generic bf16 WMMA GEMM: C = A[M,K] @ Bt[N,K]^T.
// Block tile 64x128, 256 threads = 8 waves (2x4), wave tile 32x32.
// TDM double-buffered LDS: wave 0 issues tensor_load_to_lds for tile i+2
// while all waves run WMMA on tile i; TENSORcnt + barrier synchronize.
// M % 64 == 0, N % 128 == 0, K % 32 == 0 (true for all our calls).
// ---------------------------------------------------------------------------
template <bool F32OUT>
__global__ __launch_bounds__(256) void gemm_kernel(const __bf16* __restrict__ A,
                                                   int lda, long strideA,
                                                   const __bf16* __restrict__ Bt,
                                                   void* __restrict__ Cv,
                                                   int ldc, long strideC,
                                                   const float* __restrict__ bias,
                                                   int K) {
  const int m0 = blockIdx.y * 64;
  const int n0 = blockIdx.x * 128;
  const long bz = blockIdx.z;
  A += bz * strideA;

  __shared__ __bf16 As[2][64 * 32];     // [buf][row][k]  4KB each
  __shared__ __bf16 Bs[2][128 * 32];    // [buf][n][k]    8KB each

  const int tid = threadIdx.x, lane = tid & 31, wid = tid >> 5;
  const int wm = wid >> 2, wn = wid & 3;
  const int half = lane >> 4, l15 = lane & 15;

  v8f acc[2][2] = {};
  const int nIter = K / 32;

#if defined(USE_TDM)
  const unsigned asB = ldsAddr(&As[0][0]);
  const unsigned bsB = ldsAddr(&Bs[0][0]);
  const char* gA = (const char*)A + (size_t)m0 * lda * 2;
  const char* gB = (const char*)Bt + (size_t)n0 * K * 2;
  if (wid == 0) {
    tdm_load_2d(asB, gA, 32, 64, (unsigned long long)lda);
    tdm_load_2d(bsB, gB, 32, 128, (unsigned long long)K);
    if (nIter > 1) {
      tdm_load_2d(asB + 4096, gA + 64, 32, 64, (unsigned long long)lda);
      tdm_load_2d(bsB + 8192, gB + 64, 32, 128, (unsigned long long)K);
    }
  }
#else
  const int ra = tid >> 2, ca = tid & 3;        // A: 256 chunks of 16B
  const int rb = tid >> 1, cb = (tid & 1) * 2;  // B: 512 chunks, 2 per thread
#endif

  for (int it = 0; it < nIter; ++it) {
    const int buf = it & 1;
#if defined(USE_TDM)
    if (wid == 0) {
      if (it + 2 <= nIter) __builtin_amdgcn_s_wait_tensorcnt(2);
      else                 __builtin_amdgcn_s_wait_tensorcnt(0);
    }
    __syncthreads();    // publish TDM-filled buffer to all waves
#else
    {
      const int k0 = it * 32;
      *(v8bf*)&As[0][ra * 32 + ca * 8] =
          *(const v8bf*)&A[(size_t)(m0 + ra) * lda + k0 + ca * 8];
      *(v8bf*)&Bs[0][rb * 32 + cb * 8] =
          *(const v8bf*)&Bt[(size_t)(n0 + rb) * K + k0 + cb * 8];
      *(v8bf*)&Bs[0][rb * 32 + cb * 8 + 8] =
          *(const v8bf*)&Bt[(size_t)(n0 + rb) * K + k0 + cb * 8 + 8];
      if (k0 + 32 < K) {
        __builtin_prefetch(&A[(size_t)(m0 + ra) * lda + k0 + 32], 0, 0);
        __builtin_prefetch(&Bt[(size_t)(n0 + rb) * K + k0 + 32], 0, 0);
      }
    }
    __syncthreads();
#endif

    const __bf16* as = &As[dbuf(buf)][0];
    const __bf16* bs = &Bs[dbuf(buf)][0];
    v16bf af[2], bfr[2];
#pragma unroll
    for (int i = 0; i < 2; ++i) {
      const int row = wm * 32 + i * 16 + l15;
      af[i] = ld_frag2(&as[row * 32 + half * 8], &as[row * 32 + 16 + half * 8]);
    }
#pragma unroll
    for (int j = 0; j < 2; ++j) {
      const int row = wn * 32 + j * 16 + l15;
      const __bf16* p = &bs[row * 32 + half * 8];
      bfr[j] = ld_frag2(p, p + 16);
    }
#pragma unroll
    for (int i = 0; i < 2; ++i)
#pragma unroll
      for (int j = 0; j < 2; ++j)
        acc[i][j] = __builtin_amdgcn_wmma_f32_16x16x32_bf16(
            false, af[i], false, bfr[j], (short)0, acc[i][j], false, false);
    __syncthreads();

#if defined(USE_TDM)
    if (wid == 0 && it + 2 < nIter) {
      const size_t koff = (size_t)(it + 2) * 64;   // 32 elems * 2B
      tdm_load_2d(asB + buf * 4096, gA + koff, 32, 64, (unsigned long long)lda);
      tdm_load_2d(bsB + buf * 8192, gB + koff, 32, 128, (unsigned long long)K);
    }
#endif
  }

#pragma unroll
  for (int i = 0; i < 2; ++i)
#pragma unroll
    for (int j = 0; j < 2; ++j)
#pragma unroll
      for (int r = 0; r < 8; ++r) {
        const int Mr = m0 + wm * 32 + i * 16 + half * 8 + r;
        const int Nc = n0 + wn * 32 + j * 16 + l15;
        const size_t cidx = (size_t)(bz * strideC) + (size_t)Mr * ldc + Nc;
        if (F32OUT)
          ((float*)Cv)[cidx] = acc[i][j][r] + (bias ? bias[Nc] : 0.f);
        else
          ((__bf16*)Cv)[cidx] = (__bf16)acc[i][j][r];
      }
}

// ---------------------------------------------------------------------------
// K4: flash attention. One block = one (b,h). 128 threads = 4 waves;
// wave w owns q-rows [16w,16w+16). Streams 65 tiles of 64 keys.
// K tile is DMA'd by the TDM; V tile is transposed through VGPRs.
// ---------------------------------------------------------------------------
__global__ __launch_bounds__(128) void attn_kernel(const __bf16* __restrict__ qbuf,
                                                   const __bf16* __restrict__ kv,
                                                   const int* __restrict__ mask,
                                                   __bf16* __restrict__ attn) {
  const int b = blockIdx.x >> 4;
  const int h = blockIdx.x & 15;
  const int tid = threadIdx.x;
  const int lane = tid & 31;
  const int w = tid >> 5;
  const int half = lane >> 4;
  const int l15 = lane & 15;

  __shared__ __bf16 Ks[64 * 64];        // [f][d]
  __shared__ __bf16 Vt[64 * 64];        // [d][f]
  __shared__ __bf16 Ps[4][16 * 64];     // per-wave P tile [q][f]

  // q A-fragments (row = q, K = d), loaded once.
  const __bf16* qrow = qbuf + ((size_t)(b * Q_ + w * 16 + l15)) * HID + h * DH;
  v16bf qa[2];
#pragma unroll
  for (int kf = 0; kf < 2; ++kf) {
    const __bf16* p = qrow + kf * 32;
    qa[kf] = ld_frag2(p + half * 8, p + 16 + half * 8);
  }

  v8f o[4] = {};
  float mr[8], lr[8];
#pragma unroll
  for (int r = 0; r < 8; ++r) { mr[r] = -3.0e38f; lr[r] = 0.f; }

  const size_t kvbase = (size_t)b * FQ * 2048;
#if defined(USE_TDM)
  const unsigned ksB = ldsAddr(&Ks[0]);
#endif

  for (int t = 0; t < 65; ++t) {
    const int f0 = t * 64;
    __syncthreads();   // previous tile fully consumed before overwrite

#if defined(USE_TDM)
    if (w == 0)        // K tile: 64 rows x 64 bf16, row stride 2048 elems
      tdm_load_2d(ksB, kv + kvbase + (size_t)f0 * 2048 + h * DH, 64, 64, 2048ull);
#endif
    // V tile transposed through VGPRs (TDM cannot transpose).
#pragma unroll
    for (int i = 0; i < 4; ++i) {
      const int ci = tid + i * 128;          // 0..511
      const int fr = ci >> 3;                // key row in tile
      const int cc = ci & 7;                 // 8-elem chunk
      const __bf16* src = kv + kvbase + (size_t)(f0 + fr) * 2048 + h * DH;
#if !defined(USE_TDM)
      *(v8bf*)&Ks[fr * 64 + cc * 8] = *(const v8bf*)(src + cc * 8);
#endif
      v8bf vv = *(const v8bf*)(src + HID + cc * 8);
#pragma unroll
      for (int j = 0; j < 8; ++j) Vt[(cc * 8 + j) * 64 + fr] = vv[j];
    }
#if defined(USE_TDM)
    if (w == 0) __builtin_amdgcn_s_wait_tensorcnt(0);
#endif
    __syncthreads();

    // sim = q @ k^T  (B column f = K row f, contiguous in Ks)
    v8f s[4] = {};
#pragma unroll
    for (int nt = 0; nt < 4; ++nt)
#pragma unroll
      for (int kf = 0; kf < 2; ++kf) {
        const __bf16* p = &Ks[(nt * 16 + l15) * 64 + kf * 32 + half * 8];
        v16bf bk = ld_frag2(p, p + 16);
        s[nt] = __builtin_amdgcn_wmma_f32_16x16x32_bf16(
            false, qa[kf], false, bk, (short)0, s[nt], false, false);
      }

    // scale (dh^-0.5 = 1/8) + key mask (column-uniform per lane)
#pragma unroll
    for (int nt = 0; nt < 4; ++nt) {
      const int fc = f0 + nt * 16 + l15;
      const bool mk = (fc < F_) && (mask[b * F_ + fc] != 0);
#pragma unroll
      for (int r = 0; r < 8; ++r)
        s[nt][r] = mk ? -3.0e38f : s[nt][r] * 0.125f;
    }

    // streaming softmax update (row = q lives on 16 lanes of one half)
#pragma unroll
    for (int r = 0; r < 8; ++r) {
      float cm = fmaxf(fmaxf(s[0][r], s[1][r]), fmaxf(s[2][r], s[3][r]));
#pragma unroll
      for (int d = 1; d < 16; d <<= 1) cm = fmaxf(cm, __shfl_xor(cm, d, 32));
      const float mn = fmaxf(mr[r], cm);
      const float alpha = __expf(mr[r] - mn);
      mr[r] = mn;
      float ps = 0.f;
#pragma unroll
      for (int nt = 0; nt < 4; ++nt) {
        const float p = __expf(s[nt][r] - mn);
        s[nt][r] = p;
        ps += p;
      }
#pragma unroll
      for (int d = 1; d < 16; d <<= 1) ps += __shfl_xor(ps, d, 32);
      lr[r] = lr[r] * alpha + ps;
#pragma unroll
      for (int nt = 0; nt < 4; ++nt) o[nt][r] *= alpha;
    }

    // P -> LDS (C layout -> A layout via per-wave scratch)
#pragma unroll
    for (int nt = 0; nt < 4; ++nt)
#pragma unroll
      for (int r = 0; r < 8; ++r)
        Ps[w][(half * 8 + r) * 64 + nt * 16 + l15] = (__bf16)s[nt][r];
    __syncthreads();

    // o += P @ V   (A = P [q x f], B = Vt rows are columns of V)
#pragma unroll
    for (int kf = 0; kf < 2; ++kf) {
      const __bf16* p = &Ps[w][l15 * 64 + kf * 32];
      v16bf ap = ld_frag2(p + half * 8, p + 16 + half * 8);
#pragma unroll
      for (int nt = 0; nt < 4; ++nt) {
        const __bf16* pv = &Vt[(nt * 16 + l15) * 64 + kf * 32 + half * 8];
        v16bf bv = ld_frag2(pv, pv + 16);
        o[nt] = __builtin_amdgcn_wmma_f32_16x16x32_bf16(
            false, ap, false, bv, (short)0, o[nt], false, false);
      }
    }
  }

  // normalize + store attn[b][q][h*64+d] (bf16)
#pragma unroll
  for (int r = 0; r < 8; ++r) {
    const float inv = 1.0f / lr[r];
    const int q = w * 16 + half * 8 + r;
#pragma unroll
    for (int nt = 0; nt < 4; ++nt) {
      const int d = nt * 16 + l15;
      attn[((size_t)(b * Q_ + q)) * HID + h * DH + d] = (__bf16)(o[nt][r] * inv);
    }
  }
}

// ---------------------------------------------------------------------------
// Host launcher
// ---------------------------------------------------------------------------
extern "C" void kernel_launch(void* const* d_in, const int* in_sizes, int n_in,
                              void* d_out, int out_size, void* d_ws, size_t ws_size,
                              hipStream_t stream) {
  (void)in_sizes; (void)n_in; (void)out_size; (void)ws_size;

  const float* x    = (const float*)d_in[0];
  const float* lat  = (const float*)d_in[1];
  const int*   mask = (const int*)d_in[2];
  const float* Wq   = (const float*)d_in[3];
  const float* Wkv  = (const float*)d_in[4];
  const float* Wout = (const float*)d_in[5];
  const float* bout = (const float*)d_in[6];
  const float* gm   = (const float*)d_in[7];
  const float* bm   = (const float*)d_in[8];
  const float* gl   = (const float*)d_in[9];
  const float* bl   = (const float*)d_in[10];
  float* out = (float*)d_out;

  // Workspace layout (bytes)
  char* ws = (char*)d_ws;
  size_t off = 0;
  __bf16* WqT   = (__bf16*)(ws + off); off += (size_t)2 * HID * DIMC;        // 2 MB
  __bf16* WkvT  = (__bf16*)(ws + off); off += (size_t)2 * 2048 * DIMC;       // 4 MB
  __bf16* WoutT = (__bf16*)(ws + off); off += (size_t)2 * DIMC * HID;        // 2 MB
  __bf16* xn    = (__bf16*)(ws + off); off += (size_t)2 * B_ * FQ * DIMC;    // 68 MB
  __bf16* kvb   = (__bf16*)(ws + off); off += (size_t)2 * B_ * FQ * 2048;    // 136 MB
  __bf16* qbuf  = (__bf16*)(ws + off); off += (size_t)2 * B_ * Q_ * HID;     // 1 MB
  __bf16* attn  = (__bf16*)(ws + off); off += (size_t)2 * B_ * Q_ * HID;     // 1 MB

  dim3 tb(32, 8);
  // K0: weight transpose/convert
  wtrans_kernel<<<dim3(HID / 32, DIMC / 32), tb, 0, stream>>>(Wq,   WqT,   HID,  DIMC);
  wtrans_kernel<<<dim3(2048 / 32, DIMC / 32), tb, 0, stream>>>(Wkv, WkvT,  2048, DIMC);
  wtrans_kernel<<<dim3(DIMC / 32, HID / 32), tb, 0, stream>>>(Wout, WoutT, DIMC, HID);

  // K1: layernorm -> xn
  ln_kernel<<<B_ * FQ, 256, 0, stream>>>(x, lat, gm, bm, gl, bl, xn);

  // K2: kv = xn @ Wkv   (M = 8*4160 contiguous, N = 2048)
  gemm_kernel<false><<<dim3(2048 / 128, (B_ * FQ) / 64, 1), 256, 0, stream>>>(
      xn, DIMC, 0L, WkvT, kvb, 2048, 0L, nullptr, DIMC);

  // K3: q = ln @ Wq   (per-batch latent rows inside xn)
  gemm_kernel<false><<<dim3(HID / 128, 1, B_), 256, 0, stream>>>(
      xn + (size_t)F_ * DIMC, DIMC, (long)FQ * DIMC, WqT,
      qbuf, HID, (long)Q_ * HID, nullptr, DIMC);

  // K4: attention
  attn_kernel<<<B_ * NH, 128, 0, stream>>>(qbuf, kvb, mask, attn);

  // K5: out = attn @ Wout + bout  (M = 8*64 contiguous, f32 output)
  gemm_kernel<true><<<dim3(DIMC / 128, (B_ * Q_) / 64, 1), 256, 0, stream>>>(
      attn, HID, 0L, WoutT, out, DIMC, 0L, bout, HID);
}
